// IDMForwardSim_34531537059886
// MI455X (gfx1250) — compile-verified, matrix operand order checked by reading, not run on hardware
//
#include <hip/hip_runtime.h>
#include <math.h>

typedef __attribute__((ext_vector_type(16))) _Float16 v16h;
typedef __attribute__((ext_vector_type(8)))  float    v8f;

#define B_TOT   65536
#define Z_DIM   128
#define T_STEPS 40
#define A_DIM   2
#define P_DIM   50
#define H_DIM   100
#define G4      400      // 4*H
#define NT      7        // n sub-tiles of 16 (pad 100 -> 112)
#define KT      4        // k tiles of 32 (K = 128: h[0..99], sdv at 100..101, zeros above)
#define NCT     28       // permuted column tiles = NT*4 gates
#define QW      448      // permuted gate width = NCT*16
#define WAVES   4
#define ROWS    64       // WAVES*16 batch rows per block
#define THREADS 128
#define DT_STEP 0.1f

// ---- dynamic LDS layout (bytes) — ~268.5 KB, legal per CDNA5 (320KB/WG) ----
#define OFF_WSW   0        // _Float16[NCT*KT*32*16] = 114688 B  (B-fragment-swizzled [lstm_r ; lstm_k rows 50,51])
#define OFF_XK    114688   // float[ROWS*QW] = 114688 B          (C-fragment-swizzled; aliased as z-stage in setup)
#define OFF_ATTW  229376   // float[112]
#define OFF_ATTB  229824   // float (padded to 64B)
#define OFF_HLDS  229888   // _Float16[WAVES*16*128] = 16384 B   (row-major h, k=100,101 carry sdv)
#define OFF_PROJ  246272   // float[ROWS*P_DIM] = 12800 B
#define OFF_TMP   259072   // float[ROWS*P_DIM] = 12800 B
#define OFF_SIDM  271872   // float[ROWS*12]    = 3072 B
#define SMEM_BYTES 274944

__device__ __forceinline__ float sigmoidf(float x) { return 1.0f / (1.0f + __expf(-x)); }

extern "C" __global__ __launch_bounds__(THREADS)
void idm_lstm_rollout(const float* __restrict__ sampled_z,
                      const float* __restrict__ idm_params,
                      const float* __restrict__ idm_s,
                      const float* __restrict__ sdv_acts,
                      const float* __restrict__ W1, const float* __restrict__ b1,
                      const float* __restrict__ W2, const float* __restrict__ b2,
                      const float* __restrict__ lstm_k, const float* __restrict__ lstm_r,
                      const float* __restrict__ lstm_b,
                      const float* __restrict__ att_w, const float* __restrict__ att_b,
                      float* __restrict__ out)
{
    extern __shared__ __attribute__((aligned(128))) char smem[];
    _Float16* wsw  = (_Float16*)(smem + OFF_WSW);
    float*    xk   = (float*)(smem + OFF_XK);
    float*    attw = (float*)(smem + OFF_ATTW);
    float*    attb = (float*)(smem + OFF_ATTB);
    _Float16* hlds = (_Float16*)(smem + OFF_HLDS);
    float*    proj = (float*)(smem + OFF_PROJ);
    float*    tmp  = (float*)(smem + OFF_TMP);
    float*    sidm = (float*)(smem + OFF_SIDM);

    const int tid  = threadIdx.x;
    const int row0 = blockIdx.x * ROWS;

    // ---------- setup phase 0: zero h state, stage sampled_z (aliases xk) ----------
    for (int i = tid; i < WAVES * 16 * 128; i += THREADS) hlds[i] = (_Float16)0.0f;
    float* zst = xk; // [ROWS][Z_DIM]
    for (int i = tid; i < ROWS * Z_DIM; i += THREADS)
        zst[i] = sampled_z[(size_t)row0 * Z_DIM + i];
    __syncthreads();

    // ---------- phase 1: h1 = relu(z@W1+b1); fill swizzled weights & small vecs ----------
    for (int i = tid; i < ROWS * P_DIM; i += THREADS) {
        int r = i / P_DIM, p = i - r * P_DIM;
        float acc = b1[p];
        for (int k = 0; k < Z_DIM; ++k) acc += zst[r * Z_DIM + k] * W1[k * P_DIM + p];
        tmp[i] = fmaxf(acc, 0.0f);
    }
    // [lstm_r ; lstm_k rows 50,51] -> f16, padded (128 x 448), permuted gate columns,
    // B-fragment order: slot = ((j*KT+kt)*32 + lane)*16 + e
    for (int s = tid; s < NCT * KT * 32 * 16; s += THREADS) {
        int frag = s >> 9, within = s & 511, lane = within >> 4, e = within & 15;
        int j = frag >> 2, kt = frag & 3;
        int m = lane & 15, hi = lane >> 4;
        int k = kt * 32 + ((e >> 3) & 1) * 16 + hi * 8 + ((e >> 1) & 3) * 2 + (e & 1);
        int nt = j >> 2, g = j & 3, n = nt * 16 + m;
        float v = 0.0f;
        if (n < H_DIM) {
            int gcol = g * H_DIM + n;
            if (k < H_DIM)            v = lstm_r[k * G4 + gcol];
            else if (k == H_DIM)      v = lstm_k[50 * G4 + gcol];   // sdv0 row
            else if (k == H_DIM + 1)  v = lstm_k[51 * G4 + gcol];   // sdv1 row
        }
        wsw[s] = (_Float16)v;
    }
    if (tid < H_DIM) attw[tid] = att_w[tid];
    if (tid == H_DIM) attb[0] = att_b[0];
    __syncthreads();

    // ---------- phase 2: proj = relu(h1@W2+b2) ----------
    for (int i = tid; i < ROWS * P_DIM; i += THREADS) {
        int r = i / P_DIM, p = i - r * P_DIM;
        float acc = b2[p];
        for (int k = 0; k < P_DIM; ++k) acc += tmp[r * P_DIM + k] * W2[k * P_DIM + p];
        proj[i] = fmaxf(acc, 0.0f);
    }
    __syncthreads();

    // ---------- phase 3: hoisted xk = lstm_b + proj @ lstm_k[0:50], stored in C-fragment order
    // slot = ((w*NCT + j)*32 + lane)*8 + r  ->  row = w*16 + 8*(lane>>4) + r , q = j*16 + (lane&15)
    for (int s = tid; s < ROWS * QW; s += THREADS) {
        int r  = s & 7;
        int ln = (s >> 3) & 31;
        int rem = s >> 8;            // w*NCT + j
        int j  = rem % NCT;
        int w  = rem / NCT;
        int row = w * 16 + 8 * (ln >> 4) + r;
        int q   = j * 16 + (ln & 15);
        int nt = (q >> 4) >> 2, g = (q >> 4) & 3, n = nt * 16 + (q & 15);
        float acc = 0.0f;
        if (n < H_DIM) {
            int gcol = g * H_DIM + n;
            acc = lstm_b[gcol];
            for (int p = 0; p < P_DIM; ++p) acc += proj[row * P_DIM + p] * lstm_k[p * G4 + gcol];
        }
        xk[s] = acc;
    }
    __syncthreads();

    // ---------- rollout ----------
    const int wave = tid >> 5;
    const int lane = tid & 31;
    const int hi   = lane >> 4;   // C/D-fragment half: rows M = r + 8*hi
    const int col  = lane & 15;
    _Float16* hrow = hlds + wave * 16 * 128;

    float cst[NT][8];             // LSTM cell state, register-resident for all 40 steps
    #pragma unroll
    for (int a = 0; a < NT; ++a)
        #pragma unroll
        for (int r = 0; r < 8; ++r) cst[a][r] = 0.0f;

    const int grow = row0 + wave * 16 + (lane & 15);
    float ego_v = 0.f, ego_x = 0.f, p_dv = 1.f, p_tg = 0.f, p_jx = 0.f, p_ma = 0.f, gden = 1.f;
    if (lane < 16) {
        ego_v = idm_s[(size_t)grow * T_STEPS * 12 + 0];
        ego_x = idm_s[(size_t)grow * T_STEPS * 12 + 3];
        p_dv = idm_params[grow * 5 + 0];
        p_tg = idm_params[grow * 5 + 1];
        p_jx = idm_params[grow * 5 + 2];
        p_ma = idm_params[grow * 5 + 3];
        float p_mi = idm_params[grow * 5 + 4];
        gden = 2.0f * sqrtf(p_ma * p_mi);
    }

    float* out_acts = out;
    float* out_atts = out + (size_t)B_TOT * T_STEPS;

    for (int t = 0; t < T_STEPS; ++t) {
        // stage this step's idm rows into LDS; sdv goes straight into the A operand (k=100,101)
        for (int i = tid; i < ROWS * 12; i += THREADS) {
            int r = i / 12, c2 = i - r * 12;
            sidm[i] = idm_s[((size_t)(row0 + r) * T_STEPS + t) * 12 + c2];
        }
        if (lane < 16) {
            const float* sp = sdv_acts + ((size_t)grow * T_STEPS + t) * A_DIM;
            hrow[(lane & 15) * 128 + H_DIM]     = (_Float16)sp[0];
            hrow[(lane & 15) * 128 + H_DIM + 1] = (_Float16)sp[1];
        }
        if (t + 1 < T_STEPS)
            __builtin_prefetch(&idm_s[((size_t)(row0 + (tid >> 1)) * T_STEPS + (t + 1)) * 12], 0, 1);
        __syncthreads();

        // assemble A fragments: [h(k=0..99) | sdv(k=100,101) | zeros]
        v16h afrag[KT];
        const _Float16* hp = hrow + (lane & 15) * 128;
        #pragma unroll
        for (int kt = 0; kt < KT; ++kt) {
            #pragma unroll
            for (int p = 0; p < 8; ++p) {
                int k0 = kt * 32 + ((p >> 2) & 1) * 16 + hi * 8 + (p & 3) * 2;
                afrag[kt][2 * p]     = hp[k0];
                afrag[kt][2 * p + 1] = hp[k0 + 1];
            }
        }

        #pragma unroll
        for (int nt = 0; nt < NT; ++nt) {
            v8f acc[4];
            #pragma unroll
            for (int g = 0; g < 4; ++g) {
                int j = nt * 4 + g;
                // C init: one contiguous fragment read of hoisted xk
                v8f c = *(const v8f*)(xk + ((size_t)(wave * NCT + j) * 32 + lane) * 8);
                #pragma unroll
                for (int kt = 0; kt < KT; ++kt) {
                    v16h b = *(const v16h*)(wsw + ((size_t)(j * KT + kt) * 32 + lane) * 16);
                    c = __builtin_amdgcn_wmma_f32_16x16x32_f16(
                            false, afrag[kt], false, b, (short)0, c, false, false);
                }
                acc[g] = c;
            }
            // gate nonlinearities — pure per-lane register math (gates aligned by column permutation)
            int n = nt * 16 + col;
            #pragma unroll
            for (int r = 0; r < 8; ++r) {
                float zi = acc[0][r], zf = acc[1][r], zg = acc[2][r], zo = acc[3][r];
                float cn = sigmoidf(zf) * cst[nt][r] + sigmoidf(zi) * tanhf(zg);
                cst[nt][r] = cn;
                float hn = tanhf(cn) * sigmoidf(zo);
                if (n < H_DIM) hrow[(hi * 8 + r) * 128 + n] = (_Float16)hn;
            }
        }

        // attention: lane L<16 sums n=0..49 of row L, partner lane sums n=50..99; combine
        {
            int m = lane & 15;
            float dot = 0.0f;
            int nbase = hi * 50;
            #pragma unroll 10
            for (int n2 = 0; n2 < 50; ++n2)
                dot += (float)hrow[m * 128 + nbase + n2] * attw[nbase + n2];
            dot += __shfl_xor(dot, 16, 32);
            float att_pre = sigmoidf(dot + attb[0]);

            if (lane < 16) {
                const float* sr = sidm + (wave * 16 + lane) * 12;
                float f_v = sr[1], m_v = sr[2], f_x = sr[4], m_x = sr[5];
                float f_ex = sr[10], m_ex = sr[11];
                float ef_dx = fminf(fmaxf(f_x - ego_x, 0.1f), 1000.0f);
                float em_dx = fminf(fmaxf(m_x - ego_x, 0.1f), 1000.0f);
                float ef_dv = ego_v - f_v;
                float em_dv = ego_v - m_v;
                auto idm = [&](float dv, float dx) -> float {
                    float gap  = p_tg * ego_v + ego_v * dv / gden;
                    float dgap = p_jx + fmaxf(gap, 0.0f);
                    float vr   = ego_v / p_dv;
                    float vr2  = vr * vr;
                    float gr   = dgap / dx;
                    float a    = p_ma * (1.0f - vr2 * vr2 - gr * gr);
                    return fminf(fmaxf(a, -100000.0f), 100000.0f);
                };
                float ef_act = idm(ef_dv, ef_dx);
                float em_act = idm(em_dv, em_dx);
                float att = (f_ex * att_pre + (1.0f - f_ex)) * m_ex;
                float act = (1.0f - att) * ef_act + att * em_act;
                ego_v = ego_v + act * DT_STEP;
                ego_x = ego_x + ego_v * DT_STEP + 0.5f * act * DT_STEP * DT_STEP;
                out_acts[(size_t)grow * T_STEPS + t] = act;
                out_atts[(size_t)grow * T_STEPS + t] = att;
            }
        }
        __syncthreads();
    }
}

extern "C" void kernel_launch(void* const* d_in, const int* in_sizes, int n_in,
                              void* d_out, int out_size, void* d_ws, size_t ws_size,
                              hipStream_t stream) {
    (void)in_sizes; (void)n_in; (void)out_size; (void)d_ws; (void)ws_size;
    const float* sampled_z  = (const float*)d_in[0];
    const float* idm_params = (const float*)d_in[1];
    const float* idm_s      = (const float*)d_in[2];
    const float* sdv_acts   = (const float*)d_in[3];
    const float* W1 = (const float*)d_in[4];
    const float* b1 = (const float*)d_in[5];
    const float* W2 = (const float*)d_in[6];
    const float* b2 = (const float*)d_in[7];
    const float* lstm_k = (const float*)d_in[8];
    const float* lstm_r = (const float*)d_in[9];
    const float* lstm_b = (const float*)d_in[10];
    const float* att_w  = (const float*)d_in[11];
    const float* att_b  = (const float*)d_in[12];
    float* outp = (float*)d_out;

    dim3 grid(B_TOT / ROWS), block(THREADS);
    idm_lstm_rollout<<<grid, block, SMEM_BYTES, stream>>>(
        sampled_z, idm_params, idm_s, sdv_acts,
        W1, b1, W2, b2, lstm_k, lstm_r, lstm_b, att_w, att_b, outp);
}
